// SPGNNLayer_70866960384358
// MI455X (gfx1250) — matrix operand order repaired and checked by reading, not compiled, not run on hardware
//
#include <hip/hip_runtime.h>

// ---------------------------------------------------------------------------
// SPGNN layer for MI455X (gfx1250), wave32 + WMMA.
//   x1 = relu(relu(x@Wn1+bn1)@Wn2+bn2)          (WMMA f16->f32)
//   Wx = spmm(K)  @ x1                           (wave-per-nnz, f32 atomics)
//   x2 = spmm(A)  @ Wx + relu-mlp_self(x)        (WMMA + atomics)
//   x3 = x2 @ Wc + bc  -> 256x256 Sinkhorn (20 it, log-domain) in 256KB LDS
//   out = concat(x2[:, :63], exp(sinkhorn))      (n x 64 f32)
// Feature dim padded 63 -> 64 everywhere (padded column is exactly zero).
//
// v2: weights + hidden activations are stored in LDS in *fragment-major*
// order, so every WMMA A/B operand is one aligned 32B LDS load per lane
// (ds_load_b128 x2) instead of 16 ds_load_u16 + v_mov_b16 packing.
// ---------------------------------------------------------------------------

typedef __attribute__((ext_vector_type(16))) _Float16 v16h;
typedef __attribute__((ext_vector_type(8)))  float    v8f;

#define NF 64                // padded feature dim
#define ROWS_PER_BLOCK 128   // 8 waves x 16-row WMMA tiles

// ---------------------------------------------------------------------------
// Fused 2-layer MLP with ReLU, all-WMMA. One wave owns a 16-row tile.
// ---------------------------------------------------------------------------
__global__ __launch_bounds__(256)
void mlp2_wmma_kernel(const float* __restrict__ X,    // n x 64
                      const float* __restrict__ W1,   // 64 x 63 row-major
                      const float* __restrict__ B1,   // 63
                      const float* __restrict__ W2,   // 63 x 63 row-major
                      const float* __restrict__ B2,   // 63
                      float* __restrict__ OUT)        // n x 64 (col 63 == 0)
{
    // Fragment-major weight storage: index (kc32*4 + nt)*32 + lane -> v16h.
    // B-fragment layout (32x16 f16): N=lane%16, K=8*(lane/16)+(e&7)+16*(e>>3).
    __shared__ v16h sW1f[2 * 4 * 32];      // 8 KB
    __shared__ v16h sW2f[2 * 4 * 32];      // 8 KB
    __shared__ float sB1[NF];
    __shared__ float sB2[NF];
    // Hidden activations in layer-2 A-fragment order:
    // index (wave*2 + kc32)*32 + lane -> v16h.
    __shared__ v16h sHf[8 * 2 * 32];       // 16 KB

    const int t = threadIdx.x;

    // Cooperative fill of fragment-major weights (f32 -> f16, zero-padded).
    {
        _Float16* w1 = (_Float16*)sW1f;
        _Float16* w2 = (_Float16*)sW2f;
        for (int i = t; i < 2 * 4 * 32 * 16; i += 256) {
            const int e    = i & 15;
            const int ln   = (i >> 4) & 31;
            const int nt   = (i >> 9) & 3;
            const int kc32 = i >> 11;
            const int k = kc32 * 32 + (ln >> 4) * 8 + (e & 7) + (e >> 3) * 16;   // 0..63
            const int j = nt * 16 + (ln & 15);              // 0..63
            w1[i] = (_Float16)((j < 63) ? W1[k * 63 + j] : 0.0f);
            w2[i] = (_Float16)((k < 63 && j < 63) ? W2[k * 63 + j] : 0.0f);
        }
    }
    if (t < NF) {
        sB1[t] = (t < 63) ? B1[t] : 0.0f;
        sB2[t] = (t < 63) ? B2[t] : 0.0f;
    }
    __syncthreads();

    const int wave = t >> 5;            // 0..7
    const int lane = t & 31;
    const int lrow = lane & 15;         // M (A/D) or N (B/D) sub-index
    const int hi   = lane >> 4;         // lane half selector
    const int rowBase = blockIdx.x * ROWS_PER_BLOCK + wave * 16;

    // ---------------- layer 1: H = relu(X @ W1 + b1) ----------------
    v8f acc[4] = {};
    #pragma unroll
    for (int kc32 = 0; kc32 < 2; ++kc32) {
        // A fragment (16x32) straight from global X (f32 -> f16 per lane).
        const float* xr = X + (size_t)(rowBase + lrow) * NF;
        const int kA = kc32 * 32 + hi * 8;
        v16h a;
        #pragma unroll
        for (int e = 0; e < 8; ++e) {
            a[e]     = (_Float16)xr[kA + e];
            a[e + 8] = (_Float16)xr[kA + 16 + e];
        }
        #pragma unroll
        for (int nt = 0; nt < 4; ++nt) {
            const v16h b = sW1f[(kc32 * 4 + nt) * 32 + lane];  // 1x32B load
            acc[nt] = __builtin_amdgcn_wmma_f32_16x16x32_f16(
                false, a, false, b, (short)0, acc[nt], false, false);
        }
    }
    // Epilogue 1: bias + ReLU -> scatter into layer-2 A-fragment positions.
    {
        _Float16* hp = (_Float16*)sHf;
        #pragma unroll
        for (int nt = 0; nt < 4; ++nt) {
            #pragma unroll
            for (int r = 0; r < 8; ++r) {
                const int m  = r + hi * 8;          // tile row
                const int nn = nt * 16 + lrow;      // hidden feature = K of L2
                float v = acc[nt][r] + sB1[nn];
                v = v > 0.0f ? v : 0.0f;
                const int kc32 = nn >> 5;
                const int ko   = nn & 31;
                const int hi2  = (ko >> 3) & 1;
                const int e    = (ko & 7) + ((ko >= 16) ? 8 : 0);
                const int frag = (wave * 2 + kc32) * 32 + (m + 16 * hi2);
                hp[frag * 16 + e] = (_Float16)v;
            }
        }
    }
    __syncthreads();

    // ---------------- layer 2: OUT = relu(H @ W2 + b2) ----------------
    v8f acc2[4] = {};
    #pragma unroll
    for (int kc32 = 0; kc32 < 2; ++kc32) {
        const v16h a = sHf[(wave * 2 + kc32) * 32 + lane];     // 1x32B load
        #pragma unroll
        for (int nt = 0; nt < 4; ++nt) {
            const v16h b = sW2f[(kc32 * 4 + nt) * 32 + lane];  // 1x32B load
            acc2[nt] = __builtin_amdgcn_wmma_f32_16x16x32_f16(
                false, a, false, b, (short)0, acc2[nt], false, false);
        }
    }
    // Epilogue 2: bias + ReLU -> global (all 64 columns; col 63 == 0).
    #pragma unroll
    for (int nt = 0; nt < 4; ++nt) {
        #pragma unroll
        for (int r = 0; r < 8; ++r) {
            const int m  = r + hi * 8;
            const int nn = nt * 16 + lrow;
            float v = acc2[nt][r] + sB2[nn];
            v = v > 0.0f ? v : 0.0f;
            OUT[(size_t)(rowBase + m) * NF + nn] = v;
        }
    }
}

// ---------------------------------------------------------------------------
// COO SpMM: out[row] += val * dense[col].  One wave per nnz; 2 features/lane.
// Reads are 2x128B coalesced; dense operand is L2-resident (16.8 MB << 192MB).
// ---------------------------------------------------------------------------
__global__ __launch_bounds__(256)
void spmm_atomic_kernel(const int* __restrict__ idx,    // (2, nnz) int32
                        const float* __restrict__ val,  // nnz
                        const float* __restrict__ dense,// n x 64
                        float* __restrict__ out,        // n x 64 (+=)
                        int nnz)
{
    const int lane = threadIdx.x & 31;
    const int wid  = blockIdx.x * (blockDim.x >> 5) + (threadIdx.x >> 5);
    const int nw   = gridDim.x * (blockDim.x >> 5);
    for (int e = wid; e < nnz; e += nw) {
        // Prefetch next gather row one iteration ahead (global_prefetch_b8).
        const int e2 = e + nw;
        if (e2 < nnz) {
            const int c2 = idx[nnz + e2];
            __builtin_prefetch(dense + (size_t)c2 * NF + lane, 0, 0);
        }
        const int   r = idx[e];
        const int   c = idx[nnz + e];
        const float v = val[e];
        const float* src = dense + (size_t)c * NF;
        float*       dst = out   + (size_t)r * NF;
        atomicAdd(&dst[lane],      v * src[lane]);
        atomicAdd(&dst[lane + 32], v * src[lane + 32]);
    }
}

// ---------------------------------------------------------------------------
// Classifier: sink[r*256 + c] = (x2[i] . Wc + bc) / TAU, with i = c*256 + r.
// (This bakes in both transposes of the reference's reshape dance.)
// ---------------------------------------------------------------------------
__global__ __launch_bounds__(256)
void classifier_kernel(const float* __restrict__ x2,  // n x 64
                       const float* __restrict__ Wc,  // 63
                       const float* __restrict__ bc,  // 1
                       float* __restrict__ sink,      // 256 x 256
                       int n)
{
    const int i = blockIdx.x * blockDim.x + threadIdx.x;
    if (i >= n) return;
    const float* row = x2 + (size_t)i * NF;
    float acc = bc[0];
    #pragma unroll
    for (int j = 0; j < 63; ++j) acc = fmaf(row[j], Wc[j], acc);
    sink[(i & 255) * 256 + (i >> 8)] = acc * 20.0f;   // 1/TAU, TAU=0.05
}

// Coalesced copy of the 63 feature channels into the 64-wide output.
__global__ __launch_bounds__(256)
void copy_feats_kernel(const float* __restrict__ x2, float* __restrict__ out,
                       int total)
{
    const int t = blockIdx.x * blockDim.x + threadIdx.x;
    if (t >= total) return;
    if ((t & 63) != 63) out[t] = x2[t];
}

// ---------------------------------------------------------------------------
// Sinkhorn, 20 log-domain iterations, single workgroup. The whole 256x256
// matrix lives in 256 KB of LDS (CDNA5: up to 320 KB per workgroup), passed
// as dynamic shared memory. Thread t owns row t (even it) / column t (odd it).
// Writes out channel 63: out[i*64+63] = exp(S[i%256][i/256]).
// ---------------------------------------------------------------------------
__global__ __launch_bounds__(256)
void sinkhorn_kernel(const float* __restrict__ sink, float* __restrict__ out)
{
    extern __shared__ float sS[];     // 256*256 floats = 256 KB
    const int t = threadIdx.x;
    for (int i = t; i < 256 * 256; i += 256) sS[i] = sink[i];
    __syncthreads();

    for (int it = 0; it < 20; ++it) {
        if ((it & 1) == 0) {
            // even: normalize over axis 2 (columns c) -> row pass, row r = t
            float m = -3.402823466e38f;
            for (int c = 0; c < 256; ++c) m = fmaxf(m, sS[t * 256 + c]);
            float s = 0.0f;
            for (int c = 0; c < 256; ++c) s += __expf(sS[t * 256 + c] - m);
            const float lse = m + __logf(s);
            for (int c = 0; c < 256; ++c) sS[t * 256 + c] -= lse;
        } else {
            // odd: normalize over axis 1 (rows r) -> column pass, col c = t
            float m = -3.402823466e38f;
            for (int r = 0; r < 256; ++r) m = fmaxf(m, sS[r * 256 + t]);
            float s = 0.0f;
            for (int r = 0; r < 256; ++r) s += __expf(sS[r * 256 + t] - m);
            const float lse = m + __logf(s);
            for (int r = 0; r < 256; ++r) sS[r * 256 + t] -= lse;
        }
        __syncthreads();
    }
    // node i = c*256 + r  ->  value exp(S[r][c]) into channel 63
    for (int i = t; i < 256 * 256; i += 256)
        out[(size_t)i * NF + 63] = __expf(sS[(i & 255) * 256 + (i >> 8)]);
}

__global__ __launch_bounds__(256)
void zero_kernel(float* __restrict__ p, int count)
{
    const int t = blockIdx.x * blockDim.x + threadIdx.x;
    if (t < count) p[t] = 0.0f;
}

// ---------------------------------------------------------------------------
extern "C" void kernel_launch(void* const* d_in, const int* in_sizes, int n_in,
                              void* d_out, int out_size, void* d_ws, size_t ws_size,
                              hipStream_t stream)
{
    const float* K_value = (const float*)d_in[0];
    const int*   K_index = (const int*)  d_in[1];
    const float* A_value = (const float*)d_in[2];
    const int*   A_index = (const int*)  d_in[3];
    const float* x       = (const float*)d_in[4];
    const float* Wn1 = (const float*)d_in[5];
    const float* bn1 = (const float*)d_in[6];
    const float* Wn2 = (const float*)d_in[7];
    const float* bn2 = (const float*)d_in[8];
    const float* Ws1 = (const float*)d_in[9];
    const float* bs1 = (const float*)d_in[10];
    const float* Ws2 = (const float*)d_in[11];
    const float* bs2 = (const float*)d_in[12];
    const float* Wc  = (const float*)d_in[13];
    const float* bc  = (const float*)d_in[14];

    const int nnzK = in_sizes[0];
    const int nnzA = in_sizes[2];
    const int n    = in_sizes[4] / NF;       // 65536

    // Workspace: x1 | Wx | x2 | sink  (3 * n*64 f32 + 256KB)
    const size_t fsz = (size_t)n * NF * sizeof(float);
    char* ws = (char*)d_ws;
    float* x1   = (float*)(ws);
    float* Wx   = (float*)(ws + fsz);
    float* x2   = (float*)(ws + 2 * fsz);
    float* sink = (float*)(ws + 3 * fsz);

    // 1) zero the SpMM-K accumulator (x2 is fully overwritten by mlp_self)
    zero_kernel<<<(n * NF + 255) / 256, 256, 0, stream>>>(Wx, n * NF);

    // 2) x1 = mlp2(x; Wn*) ; x2 = mlp2(x; Ws*)   (WMMA)
    const int mlpBlocks = n / ROWS_PER_BLOCK;
    mlp2_wmma_kernel<<<mlpBlocks, 256, 0, stream>>>(x, Wn1, bn1, Wn2, bn2, x1);
    mlp2_wmma_kernel<<<mlpBlocks, 256, 0, stream>>>(x, Ws1, bs1, Ws2, bs2, x2);

    // 3) Wx = spmm(K) @ x1 ; x2 += spmm(A) @ Wx
    spmm_atomic_kernel<<<2048, 256, 0, stream>>>(K_index, K_value, x1, Wx, nnzK);
    spmm_atomic_kernel<<<1024, 256, 0, stream>>>(A_index, A_value, Wx, x2, nnzA);

    // 4) classifier -> sink matrix; copy features into output
    classifier_kernel<<<(n + 255) / 256, 256, 0, stream>>>(x2, Wc, bc, sink, n);
    copy_feats_kernel<<<(n * NF + 255) / 256, 256, 0, stream>>>(x2, (float*)d_out, n * NF);

    // 5) Sinkhorn in one workgroup with 256KB dynamic LDS -> output channel 63
    sinkhorn_kernel<<<1, 256, 256 * 256 * sizeof(float), stream>>>(sink, (float*)d_out);
}
